// NGCF_66185446031938
// MI455X (gfx1250) — matrix operand order repaired and checked by reading, not compiled, not run on hardware
//
#include <hip/hip_runtime.h>
#include <hip/hip_bf16.h>
#include <stdint.h>

// ---------------------------------------------------------------------------
// NGCF forward for MI455X (gfx1250, wave32).
//   x = emb[x_ids]                        -> d_out[:, 0:128]
//   per layer l in 1..3:
//     h  = scatter_add(vals * x_{l-1}[col], row)        (fp32 atomics, L2)
//     u  = bf16(h + x_{l-1}),  v = bf16(h * x_{l-1})
//     z  = u @ Wa^T + v @ Wb^T + ba + bb  (V_WMMA_F32_16X16X32_BF16, f32 acc)
//     x_l = leaky_relu(z, 0.2)            -> d_out[:, l*128 : (l+1)*128]
// Workspace: h (N*128*4 B) + u,v (N*128*2 B each) + 6 bf16 weights (~103 MB).
// ---------------------------------------------------------------------------

#define DIMD 128
#define OUTSTRIDE (4 * DIMD) // 512 floats per output row
#define NEG_SLOPE 0.2f

typedef __attribute__((ext_vector_type(16))) __bf16 v16bf;
typedef __attribute__((ext_vector_type(8)))  float  v8f;
typedef __attribute__((ext_vector_type(4)))  unsigned int v4u;

union TileU { v16bf v; v4u q[2]; };

static __device__ __forceinline__ unsigned short f32_to_bf16_rne(float f) {
  unsigned int u = __float_as_uint(f);
  u += 0x7FFFu + ((u >> 16) & 1u);
  return (unsigned short)(u >> 16);
}

// ---------------- gather embedding into output column block 0 ---------------
__global__ void ngcf_gather_emb(const int* __restrict__ x_ids,
                                const float* __restrict__ emb,
                                float* __restrict__ out, int N) {
  int idx = blockIdx.x * blockDim.x + threadIdx.x;
  if (idx >= N * DIMD) return;
  int r = idx >> 7, d = idx & (DIMD - 1);
  out[(size_t)r * OUTSTRIDE + d] = emb[(size_t)x_ids[r] * DIMD + d];
}

// ---------------- pack one fp32 [128,128] weight matrix to bf16 -------------
__global__ void ngcf_pack_w(const float* __restrict__ W,
                            unsigned short* __restrict__ Wb) {
  int idx = blockIdx.x * blockDim.x + threadIdx.x;
  if (idx < DIMD * DIMD) Wb[idx] = f32_to_bf16_rne(W[idx]);
}

// ---------------- SPMM: one edge per wave32, float4 per lane ----------------
__global__ void ngcf_spmm(const int* __restrict__ row,
                          const int* __restrict__ col,
                          const float* __restrict__ vals,
                          const float* __restrict__ xprev, // row stride OUTSTRIDE
                          float* __restrict__ h, int E) {
  int lane   = threadIdx.x & 31;
  int wave   = (blockIdx.x * blockDim.x + threadIdx.x) >> 5;
  int nwaves = (gridDim.x * blockDim.x) >> 5;
  for (int e = wave; e < E; e += nwaves) {
    int   r = row[e];
    int   c = col[e];
    float s = vals[e];
    const float4* src = (const float4*)(xprev + (size_t)c * OUTSTRIDE) + lane;
    float4 xv = *src; // coalesced 512B gather per wave; x fits in 192MB L2
    float* dst = h + (size_t)r * DIMD + lane * 4;
    atomicAdd(dst + 0, s * xv.x);
    atomicAdd(dst + 1, s * xv.y);
    atomicAdd(dst + 2, s * xv.z);
    atomicAdd(dst + 3, s * xv.w);
  }
}

// ---------------- elementwise: u = bf16(h + x), v = bf16(h * x) -------------
__global__ void ngcf_uv(const float* __restrict__ h,
                        const float* __restrict__ xprev, // row stride OUTSTRIDE
                        unsigned short* __restrict__ u,
                        unsigned short* __restrict__ v, int N) {
  int idx = blockIdx.x * blockDim.x + threadIdx.x;
  if (idx >= N * DIMD) return;
  int r = idx >> 7, d = idx & (DIMD - 1);
  float x  = xprev[(size_t)r * OUTSTRIDE + d];
  float hh = h[idx];
  u[idx] = f32_to_bf16_rne(hh + x);
  v[idx] = f32_to_bf16_rne(hh * x);
}

// A-matrix (16x32 bf16) per-lane load per ISA layout:
// lane<16: M=lane, v0-3 hold K 0..7, v4-7 hold K 16..23 (hi half: +8 / +24)
static __device__ __forceinline__ v16bf load_a(const unsigned short* base,
                                               int sub, int hi, int kk) {
  const unsigned short* p = base + (size_t)sub * DIMD + kk + hi * 8;
  TileU t;
  t.q[0] = *(const v4u*)p;
  t.q[1] = *(const v4u*)(p + 16);
  return t.v;
}

// B-matrix (32x16 bf16) per-lane load per ISA layout:
// lane<16: N=lane, v0-3 hold K 0..7, v4-7 hold K 16..23 (hi half: +8 / +24)
// We need B[k][j] = W[j][k]; W row-major row j supplies lane j's K-run directly.
static __device__ __forceinline__ v16bf load_b(const unsigned short* wjbase,
                                               int sub, int hi, int kk) {
  const unsigned short* p = wjbase + (size_t)sub * DIMD + kk + hi * 8;
  TileU t;
  t.q[0] = *(const v4u*)p;
  t.q[1] = *(const v4u*)(p + 16);
  return t.v;
}

// ---------------- fused dual-WMMA GEMM + bias + leaky-relu ------------------
// Each wave: 16 output rows x 128 cols (8 tiles). 64 WMMAs / wave.
__global__ void __launch_bounds__(256, 2)
ngcf_gemm(const unsigned short* __restrict__ u,
          const unsigned short* __restrict__ v,
          const unsigned short* __restrict__ Wa,
          const unsigned short* __restrict__ Wb,
          const float* __restrict__ ba, const float* __restrict__ bb,
          float* __restrict__ zout, // row stride OUTSTRIDE, pre-offset column
          int N) {
  int lane = threadIdx.x & 31;
  int wave = threadIdx.x >> 5;
  int tile = blockIdx.x * 8 + wave; // 16-row tile (N % 16 == 0 here)
  if (tile >= (N >> 4)) return;
  int sub = lane & 15, hi = lane >> 4;

  v8f acc[8];
#pragma unroll
  for (int j = 0; j < 8; ++j) acc[j] = (v8f){0.f,0.f,0.f,0.f,0.f,0.f,0.f,0.f};

  const unsigned short* uBase = u + (size_t)tile * 16 * DIMD;
  const unsigned short* vBase = v + (size_t)tile * 16 * DIMD;

#pragma unroll
  for (int kk = 0; kk < DIMD; kk += 32) {
    v16bf au = load_a(uBase, sub, hi, kk);
    v16bf av = load_a(vBase, sub, hi, kk);
#pragma unroll
    for (int j = 0; j < 8; ++j) {
      v16bf bwa = load_b(Wa + (size_t)j * 16 * DIMD, sub, hi, kk);
      v16bf bwb = load_b(Wb + (size_t)j * 16 * DIMD, sub, hi, kk);
      acc[j] = __builtin_amdgcn_wmma_f32_16x16x32_bf16(
          false, au, false, bwa, (short)0, acc[j], false, false);
      acc[j] = __builtin_amdgcn_wmma_f32_16x16x32_bf16(
          false, av, false, bwb, (short)0, acc[j], false, false);
    }
  }

  // C/D layout: vgpr r, lane<16 -> row r, col lane; lane>=16 -> row r+8
  int row0 = tile * 16;
#pragma unroll
  for (int j = 0; j < 8; ++j) {
    int colj  = j * 16 + sub;
    float bias = ba[colj] + bb[colj];
#pragma unroll
    for (int r = 0; r < 8; ++r) {
      float w = acc[j][r] + bias;
      w = (w >= 0.f) ? w : NEG_SLOPE * w;
      zout[(size_t)(row0 + r + hi * 8) * OUTSTRIDE + colj] = w;
    }
  }
}

extern "C" void kernel_launch(void* const* d_in, const int* in_sizes, int n_in,
                              void* d_out, int out_size, void* d_ws, size_t ws_size,
                              hipStream_t stream) {
  (void)n_in; (void)out_size; (void)ws_size;
  const int*   x_ids = (const int*)d_in[0];
  const int*   row   = (const int*)d_in[1];
  const int*   col   = (const int*)d_in[2];
  const float* vals  = (const float*)d_in[3];
  const float* emb   = (const float*)d_in[4];
  const float* W[6];
  const float* B[6];
  for (int i = 0; i < 6; ++i) {
    W[i] = (const float*)d_in[5 + 2 * i];
    B[i] = (const float*)d_in[6 + 2 * i];
  }
  const int N = in_sizes[0];
  const int E = in_sizes[1];
  float* out = (float*)d_out;

  // workspace layout
  char* ws = (char*)d_ws;
  float*          h   = (float*)ws;                                 // N*D fp32
  unsigned short* u   = (unsigned short*)(ws + (size_t)N * DIMD * 4);
  unsigned short* v   = u + (size_t)N * DIMD;
  unsigned short* Wbf = v + (size_t)N * DIMD;                       // 6 * D*D

  const int elemBlocks = (N * DIMD + 255) / 256;
  ngcf_gather_emb<<<elemBlocks, 256, 0, stream>>>(x_ids, emb, out, N);
  for (int i = 0; i < 6; ++i)
    ngcf_pack_w<<<(DIMD * DIMD + 255) / 256, 256, 0, stream>>>(
        W[i], Wbf + (size_t)i * DIMD * DIMD);

  for (int l = 0; l < 3; ++l) {
    const float* xprev = out + l * DIMD;       // column block l, stride 512
    float*       znew  = out + (l + 1) * DIMD; // column block l+1
    hipMemsetAsync(h, 0, (size_t)N * DIMD * sizeof(float), stream);
    ngcf_spmm<<<4096, 256, 0, stream>>>(row, col, vals, xprev, h, E);
    ngcf_uv<<<elemBlocks, 256, 0, stream>>>(h, xprev, u, v, N);
    const int ntiles = N / 16;
    const int gblocks = (ntiles + 7) / 8;
    ngcf_gemm<<<gblocks, 256, 0, stream>>>(
        u, v, Wbf + (size_t)(2 * l) * DIMD * DIMD,
        Wbf + (size_t)(2 * l + 1) * DIMD * DIMD, B[2 * l], B[2 * l + 1], znew, N);
  }
}
